// PFNLayer_30142080483914
// MI455X (gfx1250) — compile-verified, matrix operand order checked by reading, not run on hardware
//
#include <hip/hip_runtime.h>
#include <hip/hip_bf16.h>

typedef __attribute__((ext_vector_type(16))) __bf16 v16bf;
typedef __attribute__((ext_vector_type(8)))  float  v8f;

#define IN_FEAT  320
#define OUT_FEAT 2048
#define KDIM     3200   // 320*9 (silu + 8 spline bases) + 320 (block-diag linear)
#define FSTRIDE  1608   // LDS feature row stride (bf16): 3216B = 804 dwords (=4 mod 8) -> conflict-free
#define MTILE    48     // 3 M-tiles: 384 accum VGPRs, fits the 512-VGPR/wave ceiling
#define NMT      3
#define BLOCK    256

union FragU { v16bf v; uint4 q[2]; };

// ---------------------------------------------------------------------------
// Prep: build augmented bf16 weight matrix W[2048][3200]
// ---------------------------------------------------------------------------
__global__ void build_w_kernel(const float* __restrict__ base_w,
                               const float* __restrict__ spline_w,
                               const float* __restrict__ scaler,
                               const float* __restrict__ lin_w,
                               __bf16* __restrict__ W) {
  long idx = (long)blockIdx.x * BLOCK + threadIdx.x;
  if (idx >= (long)OUT_FEAT * KDIM) return;
  int o = (int)(idx / KDIM);
  int j = (int)(idx % KDIM);
  float v;
  if (j < 2880) {
    int i = j / 9, t = j - i * 9;
    if (t == 0) v = base_w[o * IN_FEAT + i];
    else        v = spline_w[(o * IN_FEAT + i) * 8 + (t - 1)] * scaler[o * IN_FEAT + i];
  } else {
    int i = j - 2880;
    int p = i / 10, ii = i - p * 10;
    v = ((o >> 6) == p) ? lin_w[(o & 63) * 10 + ii] : 0.0f;
  }
  W[idx] = (__bf16)v;
}

// ---------------------------------------------------------------------------
// One k-step: 3 A fragments (LDS) x 16 B fragments (global/L2) -> 48 WMMAs
// ---------------------------------------------------------------------------
__device__ __forceinline__ void gemm_step(const __bf16* feat, const __bf16* wbase,
                                          int klocal, int kglob, int ln15, int koff,
                                          v8f acc[NMT][16]) {
  FragU a[NMT];
#pragma unroll
  for (int mt = 0; mt < NMT; ++mt) {
    const uint4* pa = (const uint4*)(feat + (size_t)(mt * 16 + ln15) * FSTRIDE + klocal + koff);
    a[mt].q[0] = pa[0]; a[mt].q[1] = pa[2];
  }
#pragma unroll
  for (int t = 0; t < 16; ++t) {
    FragU b;
    const uint4* pb = (const uint4*)(wbase + (size_t)t * 16 * KDIM + kglob);
    b.q[0] = pb[0]; b.q[1] = pb[2];
#pragma unroll
    for (int mt = 0; mt < NMT; ++mt)
      acc[mt][t] = __builtin_amdgcn_wmma_f32_16x16x32_bf16(
          false, a[mt].v, false, b.v, (short)0, acc[mt][t], false, false);
  }
}

// ---------------------------------------------------------------------------
// Fused kernel: closed-form features -> bf16 WMMA GEMM (K=3200, two chunks) ->
// BN/ReLU/point-max/concat. Workgroup = 48 samples x all 2048 outputs.
// ---------------------------------------------------------------------------
__global__ __launch_bounds__(BLOCK)
void pfn_kan_kernel(const float* __restrict__ x,      // (N, 320)
                    const __bf16* __restrict__ W,     // (2048, 3200) bf16
                    const float* __restrict__ gamma,
                    const float* __restrict__ beta,
                    const float* __restrict__ mean,
                    const float* __restrict__ var,
                    float* __restrict__ out,          // (N, 32, 128)
                    int Ntot) {
  extern __shared__ char smem[];
  __bf16* feat    = (__bf16*)smem;                                           // 48*FSTRIDE bf16
  int*    maxbuf  = (int*)(smem + (size_t)MTILE * FSTRIDE * sizeof(__bf16)); // 48*64 int
  float*  bnscale = (float*)((char*)maxbuf + MTILE * 64 * sizeof(int));      // 64
  float*  bnshift = bnscale + 64;                                            // 64

  const int tid = threadIdx.x;
  const int n0  = blockIdx.x * MTILE;

  for (int i = tid; i < MTILE * 64; i += BLOCK) maxbuf[i] = 0;
  if (tid < 64) {
    float sc = gamma[tid] * rsqrtf(var[tid] + 1e-3f);
    bnscale[tid] = sc;
    bnshift[tid] = beta[tid] - mean[tid] * sc;
  }

  const int lane = tid & 31;
  const int wv   = tid >> 5;
  const int ln15 = lane & 15;
  const int hi   = lane >> 4;
  const int koff = hi * 8;

  v8f acc[NMT][16];
  v8f zero = {};
#pragma unroll
  for (int mt = 0; mt < NMT; ++mt)
#pragma unroll
    for (int t = 0; t < 16; ++t) acc[mt][t] = zero;

  const __bf16* wbase = W + (size_t)(wv * 256 + ln15) * KDIM + koff;

  for (int kc = 0; kc < 2; ++kc) {
    __syncthreads();   // previous chunk fully consumed (and init visible)

    // ---- features for inputs i in [kc*160, kc*160+160), uniform-grid closed form ----
    for (int e = tid; e < MTILE * 160; e += BLOCK) {
      int s = e / 160, ii = e - s * 160;
      int i = kc * 160 + ii;
      int row = n0 + s; row = row < Ntot ? row : Ntot - 1;     // clamp partial tile
      float xv = __builtin_nontemporal_load(&x[(size_t)row * IN_FEAT + i]);
      float silu = xv / (1.0f + __expf(-xv));

      // interval index m in grid [-2.2, 2.2), h = 0.4; 4 nonzero cubic weights
      float tpos = (xv + 2.2f) * 2.5f;
      float mf = floorf(tpos);
      float u  = tpos - mf;
      int   m  = (int)mf;
      m = m < -4 ? -4 : (m > 14 ? 14 : m);
      float um = 1.0f - u;
      float u2 = u * u, u3 = u2 * u;
      float wa = um * um * um * (1.0f / 6.0f);                         // basis j = m-3
      float wb = (3.0f * u3 - 6.0f * u2 + 4.0f) * (1.0f / 6.0f);       // basis j = m-2
      float wc = (-3.0f * u3 + 3.0f * u2 + 3.0f * u + 1.0f) * (1.0f / 6.0f); // j = m-1
      float wd = u3 * (1.0f / 6.0f);                                   // basis j = m

      __bf16* frow = feat + (size_t)s * FSTRIDE;
      frow[ii * 9 + 0] = (__bf16)silu;
#pragma unroll
      for (int j = 0; j < 8; ++j) {                 // scatter 4 weights into 8 slots
        float v = 0.0f;
        v = (j == m - 3) ? wa : v;
        v = (j == m - 2) ? wb : v;
        v = (j == m - 1) ? wc : v;
        v = (j == m)     ? wd : v;
        frow[ii * 9 + 1 + j] = (__bf16)v;
      }
      frow[1440 + ii] = (__bf16)xv;                 // linear-path feature
    }
    __syncthreads();

    // ---- GEMM over this chunk: spline cols then linear cols ----
    for (int kl = 0; kl < 1440; kl += 32)
      gemm_step(feat, wbase, kl, kc * 1440 + kl, ln15, koff, acc);
    for (int kl = 0; kl < 160; kl += 32)
      gemm_step(feat, wbase, 1440 + kl, 2880 + kc * 160 + kl, ln15, koff, acc);
  }

  // ---- Epilogue: BN + ReLU + store + point-max (4 p-values reduced in-reg) ----
#pragma unroll
  for (int mt = 0; mt < NMT; ++mt) {
#pragma unroll
    for (int j = 0; j < 4; ++j) {
      int c = j * 16 + ln15;
      float sc = bnscale[c], sh = bnshift[c];
#pragma unroll
      for (int r = 0; r < 8; ++r) {
        int s = mt * 16 + hi * 8 + r;          // C layout: VGPR r = M r (+8 for high lanes)
        if (n0 + s < Ntot) {
          float vmax = 0.0f;
#pragma unroll
          for (int tt = 0; tt < 4; ++tt) {
            int t = j + 4 * tt;
            int p = wv * 4 + tt;
            float v = fmaxf(acc[mt][t][r] * sc + sh, 0.0f);
            __builtin_nontemporal_store(v, &out[((size_t)(n0 + s) * 32 + p) * 128 + c]);
            vmax = fmaxf(vmax, v);
          }
          atomicMax(&maxbuf[s * 64 + c], __float_as_int(vmax)); // v>=0: int order == float order
        }
      }
    }
  }
  __syncthreads();

  // ---- Replicated-max half of the concat ----
  for (int i = tid; i < MTILE * 32 * 64; i += BLOCK) {
    int s = i >> 11, rem = i & 2047;
    int p = rem >> 6, c = rem & 63;
    if (n0 + s < Ntot)
      __builtin_nontemporal_store(__int_as_float(maxbuf[s * 64 + c]),
                                  &out[((size_t)(n0 + s) * 32 + p) * 128 + 64 + c]);
  }
}

// ---------------------------------------------------------------------------
extern "C" void kernel_launch(void* const* d_in, const int* in_sizes, int n_in,
                              void* d_out, int out_size, void* d_ws, size_t ws_size,
                              hipStream_t stream) {
  const float* inputs   = (const float*)d_in[0];
  const float* linear_w = (const float*)d_in[1];
  const float* base_w   = (const float*)d_in[2];
  const float* spline_w = (const float*)d_in[3];
  const float* scaler   = (const float*)d_in[4];
  const float* gamma    = (const float*)d_in[5];
  const float* beta     = (const float*)d_in[6];
  const float* mean     = (const float*)d_in[7];
  const float* var      = (const float*)d_in[8];
  float* out = (float*)d_out;
  __bf16* W  = (__bf16*)d_ws;    // 2048*3200*2 = 13.1 MB scratch

  int N = in_sizes[0] / IN_FEAT; // 40000

  long totalW = (long)OUT_FEAT * KDIM;
  build_w_kernel<<<(int)((totalW + BLOCK - 1) / BLOCK), BLOCK, 0, stream>>>(
      base_w, spline_w, scaler, linear_w, W);

  size_t shmem = (size_t)MTILE * FSTRIDE * sizeof(__bf16)   // 154368 B features
               + (size_t)MTILE * 64 * sizeof(int)           // 12288 B max buffer
               + 128 * sizeof(float);                       // BN scale/shift
  hipFuncSetAttribute((const void*)pfn_kan_kernel,
                      hipFuncAttributeMaxDynamicSharedMemorySize, (int)shmem);
  int grid = (N + MTILE - 1) / MTILE;                       // 834 (last tile partial)
  pfn_kan_kernel<<<grid, BLOCK, shmem, stream>>>(
      inputs, W, gamma, beta, mean, var, out, N);
}
